// MultiHeadSelfAttention_90821378441515
// MI455X (gfx1250) — compile-verified, matrix operand order checked by reading, not run on hardware
//
#include <hip/hip_runtime.h>

// Problem constants (from reference): B=4, S=1024, D=1024, H=16, HD=64
#define B_  4
#define S_  1024
#define D_  1024
#define H_  16
#define HD_ 64
#define D3_ 3072

typedef __attribute__((ext_vector_type(16))) _Float16 v16h;
typedef __attribute__((ext_vector_type(8)))  _Float16 h8;
typedef __attribute__((ext_vector_type(4)))  _Float16 h4;
typedef __attribute__((ext_vector_type(8)))  float    v8f;

// ---------------------------------------------------------------------------
// WMMA 16x16x32 f16 fragment loader (wave32): lane L reads row (L&15) at
// column offsets kb+[0..7] and kb+16+[0..7], kb=(L>>4)*8. Two 16-byte loads
// map straight onto the 8-VGPR operand. Works for global and LDS pointers.
// ---------------------------------------------------------------------------
__device__ __forceinline__ v16h frag_ld(const _Float16* __restrict__ p, int kb) {
  h8 lo = *(const h8*)(p + kb);
  h8 hi = *(const h8*)(p + kb + 16);
  v16h r;
#pragma unroll
  for (int i = 0; i < 8; ++i) { r[i] = lo[i]; r[i + 8] = hi[i]; }
  return r;
}

__device__ __forceinline__ v8f wmma16(v16h a, v16h b, v8f c) {
  return __builtin_amdgcn_wmma_f32_16x16x32_f16(
      false, a, false, b, (short)0, c, false, false);
}

// CDNA5 async global->LDS copy (ASYNCcnt-tracked, bypasses VGPRs).
__device__ __forceinline__ void async_b128(unsigned lds_off,
                                           const _Float16* gaddr) {
  asm volatile("global_load_async_to_lds_b128 %0, %1, off"
               :: "v"(lds_off), "v"(gaddr) : "memory");
}
__device__ __forceinline__ void wait_async4() {
  asm volatile("s_wait_asynccnt 0x4" ::: "memory");
}
__device__ __forceinline__ void wait_async0() {
  asm volatile("s_wait_asynccnt 0x0" ::: "memory");
}

// ---------------------------------------------------------------------------
// f32 -> f16 conversion (vectorized x4)
// ---------------------------------------------------------------------------
__global__ void cvt_f32_f16(const float* __restrict__ in,
                            _Float16* __restrict__ out, int n4) {
  int i = blockIdx.x * blockDim.x + threadIdx.x;
  if (i < n4) {
    float4 v = ((const float4*)in)[i];
    h4 o = { (_Float16)v.x, (_Float16)v.y, (_Float16)v.z, (_Float16)v.w };
    ((h4*)out)[i] = o;
  }
}

// ---------------------------------------------------------------------------
// GEMM: C[M=4096, N] = A[M,1024] * Bw[N,1024]^T + bias.
// 256 threads (8 waves) per 128x128 tile; per 32-K step stage A+B (16 KB)
// into double-buffered LDS with async b128 loads (4/thread/step); each wave
// runs 8 WMMAs on a 32x64 sub-tile. Rows padded 32->40 f16 => fragment
// ds_load_b128 pairs hit all 64 banks exactly once. K loop fully unrolled
// (last step peeled) so buffers/offsets are static and acc has no phis.
// ---------------------------------------------------------------------------
#define BK   32
#define LDSB 20480u  // bytes per buffer: 2 * 128 * 40 * 2

__global__ void __launch_bounds__(256) gemm_wmma(
    const _Float16* __restrict__ A, const _Float16* __restrict__ Bw,
    const float* __restrict__ bias, _Float16* __restrict__ Ch,
    float* __restrict__ Cf, int N) {
  __shared__ _Float16 smem[2][2][128][40];   // [buf][A/B][row][k(+pad)]

  const int tid  = threadIdx.x;
  const int lane = tid & 31;
  const int wv   = tid >> 5;
  const int lq   = lane & 15;
  const int kb   = (lane >> 4) << 3;
  const int wm   = (wv >> 1) * 32;
  const int wn   = (wv & 1) * 64;
  const int mblk = blockIdx.y * 128;
  const int nblk = blockIdx.x * 128;

  // producer: thread stages rows (tid>>2) and (tid>>2)+64 of both A and B
  const int prow = tid >> 2;
  const int pq   = (tid & 3) * 8;
  const _Float16* ga0 = A  + (size_t)(mblk + prow) * D_ + pq;
  const _Float16* ga1 = ga0 + (size_t)64 * D_;
  const _Float16* gb0 = Bw + (size_t)(nblk + prow) * D_ + pq;
  const _Float16* gb1 = gb0 + (size_t)64 * D_;
  const unsigned la0 = (unsigned)(uintptr_t)&smem[0][0][prow][pq];
  const unsigned la1 = (unsigned)(uintptr_t)&smem[0][0][prow + 64][pq];
  const unsigned lb0 = (unsigned)(uintptr_t)&smem[0][1][prow][pq];
  const unsigned lb1 = (unsigned)(uintptr_t)&smem[0][1][prow + 64][pq];

  auto issue = [&](int kstep, int buf) {
    size_t off = (size_t)kstep * BK;
    unsigned lo = (unsigned)buf * LDSB;
    async_b128(la0 + lo, ga0 + off);
    async_b128(la1 + lo, ga1 + off);
    async_b128(lb0 + lo, gb0 + off);
    async_b128(lb1 + lo, gb1 + off);
  };

  v8f acc[2][4] = {};
  auto consume = [&](int buf) {
    const _Float16 (*sa)[40] = smem[buf][0];
    const _Float16 (*sb)[40] = smem[buf][1];
    v16h a0 = frag_ld(&sa[wm + lq][0],      kb);
    v16h a1 = frag_ld(&sa[wm + 16 + lq][0], kb);
#pragma unroll
    for (int nt = 0; nt < 4; ++nt) {
      v16h bf = frag_ld(&sb[wn + nt * 16 + lq][0], kb);
      acc[0][nt] = wmma16(a0, bf, acc[0][nt]);
      acc[1][nt] = wmma16(a1, bf, acc[1][nt]);
    }
  };

  const int NK = D_ / BK;   // 32
  issue(0, 0);
#pragma unroll
  for (int i = 0; i < NK - 1; ++i) {
    issue(i + 1, (i + 1) & 1);
    wait_async4();          // own previous group done -> step-i tile staged
    __syncthreads();        // all waves' staging complete
    consume(i & 1);
    __syncthreads();        // done reading before buffer is overwritten
  }
  wait_async0();
  __syncthreads();
  consume((NK - 1) & 1);

  // C layout: lane L holds column n = (L&15), rows m = kb + r
  const int mbase = mblk + wm;
  const int nbase = nblk + wn;
#pragma unroll
  for (int mt = 0; mt < 2; ++mt) {
#pragma unroll
    for (int nt = 0; nt < 4; ++nt) {
      int n = nbase + nt * 16 + lq;
      float bs = bias[n];
#pragma unroll
      for (int r = 0; r < 8; ++r) {
        int m = mbase + mt * 16 + kb + r;
        float v = acc[mt][nt][r] + bs;
        if (Cf) Cf[(size_t)m * N + n] = v;
        else    Ch[(size_t)m * N + n] = (_Float16)v;
      }
    }
  }
}

// ---------------------------------------------------------------------------
// V transpose: qkv[b,s, 2048 + h*64 + d] -> vt[b,h,d,s]  (writes coalesced)
// ---------------------------------------------------------------------------
__global__ void vtrans(const _Float16* __restrict__ qkv,
                       _Float16* __restrict__ vt) {
  size_t idx = (size_t)blockIdx.x * blockDim.x + threadIdx.x;
  int s = (int)(idx & (S_ - 1));
  int d = (int)((idx >> 10) & 63);
  int h = (int)((idx >> 16) & (H_ - 1));
  int b = (int)(idx >> 20);
  vt[idx] = qkv[((size_t)(b * S_) + s) * D3_ + 2 * D_ + h * HD_ + d];
}

// ---------------------------------------------------------------------------
// Flash attention. Grid (S/64, H, B), block 128 (4 waves).
// Wave = 16 queries x full head dim, transposed-score formulation:
//   S^T[32k x 16q] = K_blk * Q^T ; O^T[64d x 16q] += V^T * P^T
// P^T's WMMA-C register layout IS the B-fragment layout (zero-shuffle reuse).
// K (32x64, pad->72) and V^T (64x32, pad->40) tiles are staged in double-
// buffered LDS via async b128 loads (4/thread/step), shared by all 4 waves.
// Outer loop manually unrolled x2 so both buffer parities are static.
// ---------------------------------------------------------------------------
__global__ void __launch_bounds__(128) attn_flash(
    const _Float16* __restrict__ qkv, const _Float16* __restrict__ vt,
    const float* __restrict__ amask, const unsigned char* __restrict__ pmask,
    _Float16* __restrict__ attn_out) {
  __shared__ _Float16 skm[2][32][72];   // K tile  [buf][key][dim(+pad)]
  __shared__ _Float16 svm[2][64][40];   // V^T tile[buf][dim][key(+pad)]

  const int tid  = threadIdx.x;
  const int lane = tid & 31;
  const int wv   = tid >> 5;
  const int lq   = lane & 15;
  const int kb   = (lane >> 4) << 3;
  const int b = blockIdx.z, h = blockIdx.y;
  const int qbase = blockIdx.x * 64 + wv * 16;

  // Q fragments: register-resident for the whole kernel
  const _Float16* qrow = qkv + ((size_t)(b * S_) + qbase + lq) * D3_ + h * HD_;
  v16h qf0 = frag_ld(qrow,      kb);
  v16h qf1 = frag_ld(qrow + 32, kb);

  const float* mrow = amask + ((size_t)b * S_ + qbase + lq) * S_;
  const unsigned char* prow = pmask + (size_t)b * S_;

  // producer: 4 async b128 per thread per 32-key step (K: 2, V^T: 2)
  const int kr = tid >> 3, kq = (tid & 7) * 8;   // K rows kr, kr+16
  const int vr = tid >> 2, vq = (tid & 3) * 8;   // V rows vr, vr+32
  const _Float16* kg = qkv + ((size_t)(b * S_) + kr) * D3_ + D_ + h * HD_ + kq;
  const _Float16* vg = vt + ((size_t)(b * H_ + h) * HD_ + vr) * S_ + vq;
  const unsigned lk0 = (unsigned)(uintptr_t)&skm[0][kr][kq];
  const unsigned lk1 = (unsigned)(uintptr_t)&skm[0][kr + 16][kq];
  const unsigned lv0 = (unsigned)(uintptr_t)&svm[0][vr][vq];
  const unsigned lv1 = (unsigned)(uintptr_t)&svm[0][vr + 32][vq];
  const unsigned skstride = (unsigned)sizeof(skm[0]);   // 4608
  const unsigned svstride = (unsigned)sizeof(svm[0]);   // 5120

  auto issue = [&](int kblk, int buf) {
    const _Float16* kgp = kg + (size_t)kblk * D3_;
    const _Float16* vgp = vg + kblk;
    unsigned ok = buf ? skstride : 0u;
    unsigned ov = buf ? svstride : 0u;
    async_b128(lk0 + ok, kgp);
    async_b128(lk1 + ok, kgp + (size_t)16 * D3_);
    async_b128(lv0 + ov, vgp);
    async_b128(lv1 + ov, vgp + (size_t)32 * S_);
  };

  v8f o[4] = {};
  float mrun = -1e30f, lrun = 0.0f;

  auto consume = [&](int kblk, int buf) {
    v8f sacc[2] = {};
#pragma unroll
    for (int t = 0; t < 2; ++t) {
      v16h kf0 = frag_ld(&skm[buf][t * 16 + lq][0],  kb);
      v16h kf1 = frag_ld(&skm[buf][t * 16 + lq][32], kb);
      sacc[t] = wmma16(kf0, qf0, sacc[t]);
      sacc[t] = wmma16(kf1, qf1, sacc[t]);
    }

    float s[2][8];
    float smax = -1e30f;
#pragma unroll
    for (int t = 0; t < 2; ++t) {
      int koff = kblk + t * 16 + kb;
      float4 m0 = *(const float4*)(mrow + koff);
      float4 m1 = *(const float4*)(mrow + koff + 4);
      unsigned long long pb = *(const unsigned long long*)(prow + koff);
      float mv[8] = { m0.x, m0.y, m0.z, m0.w, m1.x, m1.y, m1.z, m1.w };
#pragma unroll
      for (int r = 0; r < 8; ++r) {
        bool valid = ((pb >> (8 * r)) & 0xffull) != 0;
        float v = valid ? (sacc[t][r] * 0.125f + mv[r]) : -1e30f;
        s[t][r] = v;
        smax = fmaxf(smax, v);
      }
    }
    smax = fmaxf(smax, __shfl_xor(smax, 16));
    float mnew  = fmaxf(mrun, smax);
    float alpha = __expf(mrun - mnew);

    float lsum = 0.0f;
    v16h pf;
#pragma unroll
    for (int t = 0; t < 2; ++t)
#pragma unroll
      for (int r = 0; r < 8; ++r) {
        float p = __expf(s[t][r] - mnew);
        lsum += p;
        pf[t * 8 + r] = (_Float16)p;
      }
    lsum += __shfl_xor(lsum, 16);
    lrun = lrun * alpha + lsum;
    mrun = mnew;

#pragma unroll
    for (int dt = 0; dt < 4; ++dt) {
#pragma unroll
      for (int r = 0; r < 8; ++r) o[dt][r] *= alpha;
      v16h vf = frag_ld(&svm[buf][dt * 16 + lq][0], kb);
      o[dt] = wmma16(vf, pf, o[dt]);
    }
  };

  issue(0, 0);
  for (int j = 0; j < S_ / 64; ++j) {       // 16 iters, 2 sub-steps each
    const int k0 = j * 64;
    issue(k0 + 32, 1);
    wait_async4();
    __syncthreads();
    consume(k0, 0);
    __syncthreads();
    // last prefetch is a dummy re-issue (in-bounds, never consumed;
    // drained by s_endpgm's implicit wait-idle) -> branchless body
    const int knext = (j < S_ / 64 - 1) ? k0 + 64 : k0;
    issue(knext, 0);
    wait_async4();
    __syncthreads();
    consume(k0 + 32, 1);
    __syncthreads();
  }

  float invl = 1.0f / lrun;
  _Float16* orow = attn_out + ((size_t)(b * S_) + qbase + lq) * D_ + h * HD_;
#pragma unroll
  for (int dt = 0; dt < 4; ++dt) {
    h8 ov;
#pragma unroll
    for (int r = 0; r < 8; ++r) ov[r] = (_Float16)(o[dt][r] * invl);
    *(h8*)(orow + dt * 16 + kb) = ov;
  }
}

// ---------------------------------------------------------------------------
// Launcher. Inputs: x, padding_mask(bool), attention_mask, W_in, b_in, W_out, b_out
// Workspace (f16): xh 8MB | W_in 6MB | W_out 2MB | qkv 24MB | vt 8MB | attn 8MB
// ---------------------------------------------------------------------------
extern "C" void kernel_launch(void* const* d_in, const int* in_sizes, int n_in,
                              void* d_out, int out_size, void* d_ws, size_t ws_size,
                              hipStream_t stream) {
  const float*         x     = (const float*)d_in[0];
  const unsigned char* pmask = (const unsigned char*)d_in[1];
  const float*         amask = (const float*)d_in[2];
  const float*         W_in  = (const float*)d_in[3];
  const float*         b_in  = (const float*)d_in[4];
  const float*         W_out = (const float*)d_in[5];
  const float*         b_out = (const float*)d_in[6];
  float*               out   = (float*)d_out;

  char* ws = (char*)d_ws;
  _Float16* xh    = (_Float16*)ws; ws += (size_t)B_ * S_ * D_  * 2;
  _Float16* winh  = (_Float16*)ws; ws += (size_t)D3_ * D_      * 2;
  _Float16* wouth = (_Float16*)ws; ws += (size_t)D_  * D_      * 2;
  _Float16* qkvh  = (_Float16*)ws; ws += (size_t)B_ * S_ * D3_ * 2;
  _Float16* vth   = (_Float16*)ws; ws += (size_t)B_ * S_ * D_  * 2;
  _Float16* attnh = (_Float16*)ws; ws += (size_t)B_ * S_ * D_  * 2;

  {
    int nx = B_ * S_ * D_ / 4, nwi = D3_ * D_ / 4, nwo = D_ * D_ / 4;
    cvt_f32_f16<<<nx  / 256, 256, 0, stream>>>(x,     xh,    nx);
    cvt_f32_f16<<<nwi / 256, 256, 0, stream>>>(W_in,  winh,  nwi);
    cvt_f32_f16<<<nwo / 256, 256, 0, stream>>>(W_out, wouth, nwo);
  }

  // QKV projection: qkv[4096,3072] = x * W_in^T + b_in (f16 out)
  gemm_wmma<<<dim3(D3_ / 128, (B_ * S_) / 128), 256, 0, stream>>>(
      xh, winh, b_in, qkvh, nullptr, D3_);

  // V transpose -> vt[b,h,d,s]
  vtrans<<<(B_ * H_ * HD_ * S_) / 256, 256, 0, stream>>>(qkvh, vth);

  // Flash attention -> attnh[4096,1024] (f16)
  attn_flash<<<dim3(S_ / 64, H_, B_), 128, 0, stream>>>(
      qkvh, vth, amask, pmask, attnh);

  // Output projection: out[4096,1024] = attn * W_out^T + b_out (f32 out)
  gemm_wmma<<<dim3(D_ / 128, (B_ * S_) / 128), 256, 0, stream>>>(
      attnh, wouth, b_out, nullptr, out, D_);
}
